// InfoGCN_61624190763179
// MI455X (gfx1250) — compile-verified, matrix operand order checked by reading, not run on hardware
//
#include <hip/hip_runtime.h>
#include <math.h>

// ---------------------------------------------------------------------------
// InfoGCN RK4 neural-ODE for MI455X (gfx1250), wave32 + bf16 WMMA.
//
// Uses W*(X*A^T) == (W*X)*A^T to reorder each stage so every WMMA operand is
// K-major in LDS and every D tile stores as one packed contiguous write:
//   G1: u^T = x^T W^T   (LHS Xt[v][c], RHS W[o][c]; D -> U[o][v] packed)
//   G2: y   = u  A^T    (LHS U[o][v],  RHS A[v'][v]; D -> Xt[v'][o] packed)
//   w4: out^T = x^T W4^T (D -> fp32 X[c][v] packed, + bias per lane)
// Single activation buffer Xt + intermediate U, no ping-pong, one barrier
// between ops. 8 waves/block, one 16x16 D tile per wave per GEMM.
// RK4 k-state lives in registers (8 f32/thread, element i = tid*8+j of the
// [c][v]-flat 64x32 slab).  tanh is branch-free (v_exp_f32 + v_rcp_f32).
// ---------------------------------------------------------------------------

#define TT   64
#define OBS  32
#define BB   512
#define CC   64
#define VV   25
#define VP   32               // V padded to 2 WMMA tiles
#define NELEM (CC * VP)       // 2048
#define NTH  256
#define PER_TH (NELEM / NTH)  // 8

typedef __bf16 bf16;
typedef bf16  v8bf  __attribute__((ext_vector_type(8)));
typedef bf16  v16bf __attribute__((ext_vector_type(16)));
typedef float v8f   __attribute__((ext_vector_type(8)));

struct __align__(16) Smem {
  float X[NELEM];        // 8 KB  fp32 f-output, [c][v] flat (c*VP+v)
  bf16  Xt[VP * CC];     // 4 KB  activation, transposed [v][c] (v*CC+c)
  bf16  U[CC * VP];      // 4 KB  u = W*x, stored [o][v] (o*VP+v)
  bf16  W[4][CC * CC];   // 32 KB w1..w4 bf16, row-major [o][c]
  bf16  Ab[VP * VP];     // 2 KB  A zero-padded to 32x32, row-major [v'][v]
  float Bias[4][CC];     // 1 KB
  float PE[2][CC];       // 0.5 KB
};                       // ~51.5 KB

// Branch-free tanh: sign(x) * (1 - e) / (1 + e), e = exp(-2|x|).
// Lowers to v_mul + v_exp_f32 + v_rcp_f32 + adds; saturates to +-1 for
// large |x| (e underflows to 0). No EXEC-diverging range branches.
__device__ __forceinline__ float fast_tanh(float x) {
  float e = __expf(-2.f * __builtin_fabsf(x));
  float r = (1.f - e) * __builtin_amdgcn_rcpf(1.f + e);
  return __builtin_copysignf(r, x);
}

// K-major fragment load (ISA 7.12.2): lane L<16 -> row r0+L, K {0..7,16..23};
// lane L+16 -> same row, K {8..15,24..31}.  Two aligned 16B LDS loads.
__device__ __forceinline__ v16bf load_frag(const bf16* S, int ld, int r0,
                                           int k0, int lane) {
  int half = lane >> 4, r = lane & 15;
  const bf16* p = S + (r0 + r) * ld + k0 + half * 8;
  v8bf lo = *(const v8bf*)p;
  v8bf hi = *(const v8bf*)(p + 16);
  return __builtin_shufflevector(lo, hi, 0, 1, 2, 3, 4, 5, 6, 7, 8, 9, 10, 11,
                                 12, 13, 14, 15);
}

// G1: u^T(32x64) = x^T(32x64-K) * W^T; tiles m0 over v, n0 over o; K=64.
// D tile [n=o][m=v] -> U[o][v], one packed 16B store per lane.
__device__ __forceinline__ void g_wx(Smem& s, int wi) {
  int lane = threadIdx.x & 31;
  int wv = threadIdx.x >> 5;
  int m0 = (wv & 1) * 16;        // v rows (0,16)
  int n0 = (wv >> 1) * 16;       // o cols (0,16,32,48)
  v8f acc = {};
#pragma unroll
  for (int kt = 0; kt < CC; kt += 32) {
    v16bf a = load_frag(s.Xt, CC, m0, kt, lane);
    v16bf b = load_frag(s.W[wi], CC, n0, kt, lane);
    acc = __builtin_amdgcn_wmma_f32_16x16x32_bf16(false, a, false, b, (short)0,
                                                  acc, false, false);
  }
  int half = lane >> 4, r = lane & 15;
  v8bf pk;
#pragma unroll
  for (int j = 0; j < 8; ++j) pk[j] = (bf16)acc[j];
  *(v8bf*)(s.U + (n0 + r) * VP + m0 + half * 8) = pk;
}

// G2: y(64x32) = u(64x32) * A^T + b, tanh; tiles m0 over o, n0 over v'; K=32.
// D tile [n=v'][m=o] -> Xt[v'][o], one packed 16B store per lane.
__device__ __forceinline__ void g_av(Smem& s, int wi) {
  int lane = threadIdx.x & 31;
  int wv = threadIdx.x >> 5;
  int m0 = (wv >> 1) * 16;       // o rows (0,16,32,48)
  int n0 = (wv & 1) * 16;        // v' cols (0,16)
  v16bf a = load_frag(s.U, VP, m0, 0, lane);
  v16bf b = load_frag(s.Ab, VP, n0, 0, lane);
  v8f acc = {};
  acc = __builtin_amdgcn_wmma_f32_16x16x32_bf16(false, a, false, b, (short)0,
                                                acc, false, false);
  int half = lane >> 4, r = lane & 15;
  int row0 = m0 + half * 8;
  v8bf pk;
#pragma unroll
  for (int j = 0; j < 8; ++j)
    pk[j] = (bf16)fast_tanh(acc[j] + s.Bias[wi][row0 + j]);
  *(v8bf*)(s.Xt + (n0 + r) * CC + row0) = pk;
}

// Final: out^T(32x64) = x^T * W4^T + b4 (no tanh); D tile [n=o][m=v] ->
// fp32 X[o*VP + v] ([c][v] layout), one aligned 32B store per lane.
__device__ __forceinline__ void g_w4(Smem& s) {
  int lane = threadIdx.x & 31;
  int wv = threadIdx.x >> 5;
  int m0 = (wv & 1) * 16;        // v rows
  int n0 = (wv >> 1) * 16;       // o cols
  v8f acc = {};
#pragma unroll
  for (int kt = 0; kt < CC; kt += 32) {
    v16bf a = load_frag(s.Xt, CC, m0, kt, lane);
    v16bf b = load_frag(s.W[3], CC, n0, kt, lane);
    acc = __builtin_amdgcn_wmma_f32_16x16x32_bf16(false, a, false, b, (short)0,
                                                  acc, false, false);
  }
  int half = lane >> 4, r = lane & 15;
  float bias = s.Bias[3][n0 + r];
  v8f outv;
#pragma unroll
  for (int j = 0; j < 8; ++j) outv[j] = acc[j] + bias;
  *(v8f*)(s.X + (n0 + r) * VP + m0 + half * 8) = outv;
}

// Stage this thread's 8 f-input values into Xt[v][c] (c fixed per thread).
__device__ __forceinline__ void stage_in(Smem& s, const float* vals) {
  int base = threadIdx.x * PER_TH;
  int c = base >> 5;
  int v0 = base & 31;
#pragma unroll
  for (int j = 0; j < PER_TH; ++j) s.Xt[(v0 + j) * CC + c] = (bf16)vals[j];
}

// f(x): input staged in Xt. Leaves result in fp32 s.X ([c][v]).
__device__ void run_f(Smem& s) {
  __syncthreads();                       // staged input visible
  g_wx(s, 0); __syncthreads(); g_av(s, 0); __syncthreads();
  g_wx(s, 1); __syncthreads(); g_av(s, 1); __syncthreads();
  g_wx(s, 2); __syncthreads(); g_av(s, 2); __syncthreads();
  g_w4(s);    __syncthreads();
}

// One RK4 step; yreg holds this thread's 8 state elements (i = tid*8 + j).
__device__ __attribute__((noinline)) void rk4_step(Smem& s, float* yreg) {
  float k1[PER_TH], k2[PER_TH], k3[PER_TH], vals[PER_TH];
  int base = threadIdx.x * PER_TH;
  int c = base >> 5;                 // channel (constant per thread)
  float pe0 = s.PE[0][c];            // safe: caller synced after compute_pe
  float pe1 = s.PE[1][c];
  // k1 = f(y + pe0)
#pragma unroll
  for (int j = 0; j < PER_TH; ++j) vals[j] = yreg[j] + pe0;
  stage_in(s, vals);
  run_f(s);
#pragma unroll
  for (int j = 0; j < PER_TH; ++j) k1[j] = s.X[base + j];
  // k2 = f(y + k1/3 + pe0)
#pragma unroll
  for (int j = 0; j < PER_TH; ++j)
    vals[j] = yreg[j] + k1[j] * (1.f / 3.f) + pe0;
  stage_in(s, vals);
  run_f(s);
#pragma unroll
  for (int j = 0; j < PER_TH; ++j) k2[j] = s.X[base + j];
  // k3 = f(y + k2 - k1/3 + pe0)
#pragma unroll
  for (int j = 0; j < PER_TH; ++j)
    vals[j] = yreg[j] + k2[j] - k1[j] * (1.f / 3.f) + pe0;
  stage_in(s, vals);
  run_f(s);
#pragma unroll
  for (int j = 0; j < PER_TH; ++j) k3[j] = s.X[base + j];
  // k4 = f(y + k1 - k2 + k3 + pe1)
#pragma unroll
  for (int j = 0; j < PER_TH; ++j)
    vals[j] = yreg[j] + k1[j] - k2[j] + k3[j] + pe1;
  stage_in(s, vals);
  run_f(s);
  // y += (k1 + 3(k2+k3) + k4) / 8
#pragma unroll
  for (int j = 0; j < PER_TH; ++j)
    yreg[j] += (k1[j] + 3.f * (k2[j] + k3[j]) + s.X[base + j]) * 0.125f;
}

__device__ __forceinline__ void compute_pe(Smem& s, int t0, int t1) {
  int ch = threadIdx.x;
  if (ch < CC) {
    float div = __expf((float)(ch & ~1) * (-9.210340371976184f / (float)CC));
    float a0 = (float)t0 * div, a1 = (float)t1 * div;
    s.PE[0][ch] = (ch & 1) ? __cosf(a0) : __sinf(a0);
    s.PE[1][ch] = (ch & 1) ? __cosf(a1) : __sinf(a1);
  }
}

__global__ __launch_bounds__(NTH) void infogcn_ode_kernel(
    const float* __restrict__ z, const float* __restrict__ A,
    const float* __restrict__ w1, const float* __restrict__ b1,
    const float* __restrict__ w2, const float* __restrict__ b2,
    const float* __restrict__ w3, const float* __restrict__ b3,
    const float* __restrict__ w4, const float* __restrict__ b4,
    float* __restrict__ out) {
  __shared__ Smem s;
  int tid = threadIdx.x;

  // Stage parameters (bf16) into LDS.
  for (int i = tid; i < CC * CC; i += NTH) {
    s.W[0][i] = (bf16)w1[i];
    s.W[1][i] = (bf16)w2[i];
    s.W[2][i] = (bf16)w3[i];
    s.W[3][i] = (bf16)w4[i];
  }
  for (int i = tid; i < VP * VP; i += NTH) {
    int v = i >> 5, u = i & 31;  // Ab[v'][v] = A[v'][v], zero-padded
    s.Ab[i] = (v < VV && u < VV) ? (bf16)A[v * VV + u] : (bf16)0.f;
  }
  if (tid < CC) {
    s.Bias[0][tid] = b1[tid];
    s.Bias[1][tid] = b2[tid];
    s.Bias[2][tid] = b3[tid];
    s.Bias[3][tid] = b4[tid];
  }

  float yreg[PER_TH];
  int base = tid * PER_TH;
  int c = base >> 5;
  int v0 = base & 31;

  if ((int)blockIdx.x < BB) {
    // ----- sequential scan: one block per batch element, 32 RK4 steps -----
    int b = blockIdx.x;
#pragma unroll
    for (int j = 0; j < PER_TH; ++j) {
      int v = v0 + j;
      yreg[j] = (v < VV)
                    ? z[((size_t)(b * CC + c) * TT + (OBS - 1)) * VV + v]
                    : 0.f;
    }
    for (int st = 0; st < TT - OBS; ++st) {
      compute_pe(s, st, st + 1);
      __syncthreads();
      rk4_step(s, yreg);
#pragma unroll
      for (int j = 0; j < PER_TH; ++j) {
        int v = v0 + j;
        if (v < VV)
          out[((size_t)(b * CC + c) * TT + (OBS + st)) * VV + v] = yreg[j];
      }
      __syncthreads();
    }
  } else {
    // ----- z_hat: one block per (b, t) slab, single RK4 with pe[0],pe[1] --
    int idx = blockIdx.x - BB;
    int b = idx >> 5;
    int t = idx & 31;
#pragma unroll
    for (int j = 0; j < PER_TH; ++j) {
      int v = v0 + j;
      yreg[j] =
          (v < VV) ? z[((size_t)(b * CC + c) * TT + t) * VV + v] : 0.f;
    }
    compute_pe(s, 0, 1);
    __syncthreads();
    rk4_step(s, yreg);
#pragma unroll
    for (int j = 0; j < PER_TH; ++j) {
      int v = v0 + j;
      if (v < VV) out[((size_t)(b * CC + c) * TT + t) * VV + v] = yreg[j];
    }
  }
}

extern "C" void kernel_launch(void* const* d_in, const int* in_sizes, int n_in,
                              void* d_out, int out_size, void* d_ws,
                              size_t ws_size, hipStream_t stream) {
  (void)in_sizes; (void)n_in; (void)out_size; (void)d_ws; (void)ws_size;
  const float* z  = (const float*)d_in[0];
  const float* A  = (const float*)d_in[1];
  const float* w1 = (const float*)d_in[2];
  const float* b1 = (const float*)d_in[3];
  const float* w2 = (const float*)d_in[4];
  const float* b2 = (const float*)d_in[5];
  const float* w3 = (const float*)d_in[6];
  const float* b3 = (const float*)d_in[7];
  const float* w4 = (const float*)d_in[8];
  const float* b4 = (const float*)d_in[9];
  // Long-running scan blocks first so they launch immediately; the 16384
  // short z_hat blocks backfill the machine behind them.
  int nblocks = BB + BB * OBS;
  infogcn_ode_kernel<<<dim3(nblocks), dim3(NTH), 0, stream>>>(
      z, A, w1, b1, w2, b2, w3, b3, w4, b4, (float*)d_out);
}
